// GATLayer_27951647162928
// MI455X (gfx1250) — matrix operand order repaired and checked by reading, not compiled, not run on hardware
//
#include <hip/hip_runtime.h>
#include <hip/hip_bf16.h>

typedef __attribute__((ext_vector_type(16))) _Float16 v16h;
typedef __attribute__((ext_vector_type(8)))  float    v8f;

#define B_    2
#define N_    4096
#define FIN_  256
#define FOUT_ 64
#define H_    4

__device__ __forceinline__ float lrelu(float t) { return t < 0.f ? 0.2f * t : t; }

// ---------------------------------------------------------------------------
// Kernel 1: x = h @ W  (per row), write xT (f16, transposed), el, erT
// grid: B*N blocks of 64 threads (thread f computes x[b,n,f])
// ---------------------------------------------------------------------------
__global__ __launch_bounds__(64) void gat_precompute(
    const float* __restrict__ hsrc, const float* __restrict__ W,
    const float* __restrict__ Wl,   const float* __restrict__ Wr,
    _Float16* __restrict__ xT, float* __restrict__ el, float* __restrict__ erT)
{
  const int n = blockIdx.x & (N_ - 1);
  const int b = blockIdx.x >> 12;          // / N_
  const int f = threadIdx.x;

  const float* hrow = hsrc + ((size_t)b * N_ + n) * FIN_;
  float acc = 0.f;
#pragma unroll 8
  for (int k = 0; k < FIN_; ++k)
    acc = fmaf(hrow[k], W[k * FOUT_ + f], acc);

  __shared__ float sx[FOUT_];
  sx[f] = acc;
  xT[((size_t)b * FOUT_ + f) * N_ + n] = (_Float16)acc;
  __syncthreads();

  if (f < 2 * H_) {
    const int   hh  = f & (H_ - 1);
    const bool  isR = f >= H_;
    const float* Wv = isR ? Wr : Wl;
    float a = 0.f;
#pragma unroll 8
    for (int o = 0; o < FOUT_; ++o)
      a = fmaf(sx[o], Wv[hh * FOUT_ + o], a);
    if (isR) erT[((size_t)b * H_ + hh) * N_ + n] = a;
    else     el[(((size_t)b * N_) + n) * H_ + hh] = a;
  }
}

// ---------------------------------------------------------------------------
// Kernel 2: Mmax[b,h] = max_j erT[b,h,j]   (8 blocks of 256 threads)
// ---------------------------------------------------------------------------
__global__ __launch_bounds__(256) void gat_ermax(
    const float* __restrict__ erT, float* __restrict__ Mmax)
{
  const int bh = blockIdx.x;
  const float* p = erT + (size_t)bh * N_;
  float m = -3.0e38f;
  for (int i = threadIdx.x; i < N_; i += 256) m = fmaxf(m, p[i]);
  __shared__ float sm[256];
  sm[threadIdx.x] = m;
  __syncthreads();
  for (int s = 128; s > 0; s >>= 1) {
    if (threadIdx.x < s) sm[threadIdx.x] = fmaxf(sm[threadIdx.x], sm[threadIdx.x + s]);
    __syncthreads();
  }
  if (threadIdx.x == 0) Mmax[bh] = sm[0];
}

// ---------------------------------------------------------------------------
// Kernel 3: flash-style masked softmax + P@x via WMMA.
// One BLOCK = one (b, 16-row i-tile). Wave wv handles j-quarter
// [wv*1024, wv*1024+1024) for ALL 4 heads (adj read once), then waves
// exchange partials through the reused 64 KB LDS buffer and wave wv
// reduces + stores head wv. Grid: 512 blocks x 128 threads = 2048 waves.
// ---------------------------------------------------------------------------
__global__ __launch_bounds__(128) void gat_main(
    const int*      __restrict__ adj,  const float* __restrict__ el,
    const float*    __restrict__ erT,  const _Float16* __restrict__ xT,
    const float*    __restrict__ Mmax, const float* __restrict__ bias,
    float* __restrict__ out)
{
  __shared__ float smem[H_ * N_];     // 64 KB. phase 1: erT[b]; phase 2: exchange

  const int lane = threadIdx.x & 31;
  const int wv   = threadIdx.x >> 5;  // wave id == reducer head id
  const int tile = blockIdx.x;        // 0..511
  const int b    = tile >> 8;
  const int i0   = (tile & 255) << 4;

  // phase 1: cooperative LDS load of erT[b] (4*4096 floats)
  {
    const float4* src = (const float4*)(erT + (size_t)b * H_ * N_);
    float4*       dst = (float4*)smem;
    for (int idx = threadIdx.x; idx < (H_ * N_) / 4; idx += 128) dst[idx] = src[idx];
  }
  __syncthreads();

  const int m  = lane & 15;           // A-matrix row / B,C,D column
  const int hw = lane >> 4;           // half-wave (K-group select)

  const float4 elv = *(const float4*)(el + ((size_t)b * N_ + i0 + m) * H_);
  const float4 mmv = *(const float4*)(Mmax + b * H_);
  float elh[4] = {elv.x, elv.y, elv.z, elv.w};
  float mma[4] = {mmv.x, mmv.y, mmv.z, mmv.w};
  float shh[4];
#pragma unroll
  for (int h = 0; h < 4; ++h) shh[h] = lrelu(elh[h] + mma[h]);   // safe softmax shift

  const v8f zero = {0.f, 0.f, 0.f, 0.f, 0.f, 0.f, 0.f, 0.f};
  v8f acc[4][4];
#pragma unroll
  for (int h = 0; h < 4; ++h)
#pragma unroll
    for (int c = 0; c < 4; ++c) acc[h][c] = zero;
  float rsum[4] = {0.f, 0.f, 0.f, 0.f};

  const int*      adjRow = adj + (size_t)b * N_ * N_ + (size_t)(i0 + m) * N_;
  const _Float16* xTb    = xT + (size_t)b * FOUT_ * N_;

  const int j_lo = wv * (N_ / 4);
  const int j_hi = j_lo + (N_ / 4);
  for (int j0 = j_lo; j0 < j_hi; j0 += 32) {
    // adj for A-layout: element e -> K = (e<8 ? e : e+8) + 8*hw
    int av[16];
    *(int4*)&av[0]  = *(const int4*)(adjRow + j0 + 8 * hw);
    *(int4*)&av[4]  = *(const int4*)(adjRow + j0 + 8 * hw + 4);
    *(int4*)&av[8]  = *(const int4*)(adjRow + j0 + 16 + 8 * hw);
    *(int4*)&av[12] = *(const int4*)(adjRow + j0 + 16 + 8 * hw + 4);

    // B-tiles (shared across heads): element e -> K = (e<8 ? e : e+8) + 8*hw, N = m
    v16h Bt[4];
#pragma unroll
    for (int c = 0; c < 4; ++c) {
      ((int4*)&Bt[c])[0] = *(const int4*)(xTb + (size_t)(c * 16 + m) * N_ + j0 + 8 * hw);
      ((int4*)&Bt[c])[1] = *(const int4*)(xTb + (size_t)(c * 16 + m) * N_ + j0 + 16 + 8 * hw);
    }

#pragma unroll
    for (int h = 0; h < 4; ++h) {
      const float* erh = smem + h * N_ + j0;
      float ev[16];
      *(float4*)&ev[0]  = *(const float4*)(erh + 8 * hw);
      *(float4*)&ev[4]  = *(const float4*)(erh + 8 * hw + 4);
      *(float4*)&ev[8]  = *(const float4*)(erh + 16 + 8 * hw);
      *(float4*)&ev[12] = *(const float4*)(erh + 16 + 8 * hw + 4);

      v16h A;
      float s = 0.f;
#pragma unroll
      for (int e = 0; e < 16; ++e) {
        float t = lrelu(elh[h] + ev[e]);
        float p = (av[e] != 0) ? __expf(t - shh[h]) : 0.f;  // p <= 1
        s += p;
        A[e] = (_Float16)p;
      }
      rsum[h] += s;

#pragma unroll
      for (int c = 0; c < 4; ++c)
        acc[h][c] = __builtin_amdgcn_wmma_f32_16x16x32_f16(
            false, A, false, Bt[c], (short)0, acc[h][c], false, false);
    }
  }

  // ---- phase 2: cross-wave combine through reused LDS ----
  __syncthreads();                    // everyone done reading s_er region

  float* s_x = smem;                  // 4 waves * 32 lanes * 3 foreign heads * 32 = 12288 f
  float* s_r = smem + 12288;          // 4 waves * 32 lanes * 4 heads            = 512 f

#pragma unroll
  for (int h = 0; h < 4; ++h) {
    if (h != wv) {
      const int slot = h - (h > wv ? 1 : 0);
      float* p = s_x + (size_t)(((wv * 32 + lane) * 3 + slot)) * 32;
#pragma unroll
      for (int c = 0; c < 4; ++c)
#pragma unroll
        for (int r = 0; r < 8; ++r) p[c * 8 + r] = acc[h][c][r];
    }
    s_r[wv * 128 + lane * 4 + h] = rsum[h];
  }
  __syncthreads();

  // wave wv reduces head wv: own registers + 3 LDS partials
  v8f oacc[4];
#pragma unroll
  for (int c = 0; c < 4; ++c) oacc[c] = acc[wv][c];
#pragma unroll
  for (int w = 0; w < 4; ++w) {
    if (w == wv) continue;
    const int slot = wv - (wv > w ? 1 : 0);
    const float* p = s_x + (size_t)(((w * 32 + lane) * 3 + slot)) * 32;
#pragma unroll
    for (int c = 0; c < 4; ++c)
#pragma unroll
      for (int r = 0; r < 8; ++r) oacc[c][r] += p[c * 8 + r];
  }

  // full row sum for row m: 4 waves x 2 half-wave partials
  float tot = 0.f;
#pragma unroll
  for (int w = 0; w < 4; ++w)
    tot += s_r[w * 128 + m * 4 + wv] + s_r[w * 128 + (m + 16) * 4 + wv];
  const float inv = 1.0f / tot;       // empty row -> inf -> NaN (matches reference)

  float sc[8];
#pragma unroll
  for (int r = 0; r < 8; ++r)         // row of C-VGPR r is r + 8*hw
    sc[r] = __shfl(inv, r + 8 * hw, 32);

  float bb[4];
#pragma unroll
  for (int c = 0; c < 4; ++c) bb[c] = bias[c * 16 + m];

#pragma unroll
  for (int c = 0; c < 4; ++c) {
    float* op = out + (size_t)(b * N_ + i0 + 8 * hw) * (H_ * FOUT_)
                    + wv * FOUT_ + c * 16 + m;
#pragma unroll
    for (int r = 0; r < 8; ++r) {
      float v = oacc[c][r] * sc[r] + bb[c];
      v = v > 0.f ? v : (__expf(v) - 1.f);            // ELU(alpha=1)
      op[(size_t)r * (H_ * FOUT_)] = v;
    }
  }
}

// ---------------------------------------------------------------------------
extern "C" void kernel_launch(void* const* d_in, const int* in_sizes, int n_in,
                              void* d_out, int out_size, void* d_ws, size_t ws_size,
                              hipStream_t stream) {
  const int*   adj = (const int*)d_in[0];
  const float* h   = (const float*)d_in[1];
  const float* W   = (const float*)d_in[2];
  const float* Wl  = (const float*)d_in[3];
  const float* Wr  = (const float*)d_in[4];
  const float* bv  = (const float*)d_in[5];
  float* out = (float*)d_out;

  char* ws = (char*)d_ws;
  _Float16* xT  = (_Float16*)(ws);                         // B*64*4096*2  = 1 MiB
  float*    el  = (float*)(ws + (1u << 20));               // B*N*4*4      = 128 KiB
  float*    erT = (float*)(ws + (1u << 20) + (1u << 17));  // B*4*N*4      = 128 KiB
  float*    mm  = (float*)(ws + (1u << 20) + (2u << 17));  // 8 floats

  gat_precompute<<<dim3(B_ * N_), dim3(64), 0, stream>>>(h, W, Wl, Wr, xT, el, erT);
  gat_ermax<<<dim3(B_ * H_), dim3(256), 0, stream>>>(erT, mm);
  gat_main<<<dim3(B_ * N_ / 16), dim3(128), 0, stream>>>(adj, el, erT, xT, mm, bv, out);
}
